// MorphologyMemoryPool_10565619548702
// MI455X (gfx1250) — matrix operand order repaired and checked by reading, not compiled, not run on hardware
//
#include <hip/hip_runtime.h>
#include <hip/hip_bf16.h>

#define BATCH 16384
#define FDIM  1024
#define HDIM  4096
#define NPAT  50
#define NHEAD 8
#define TOPK  3

typedef __attribute__((ext_vector_type(16))) __bf16 v16bf;
typedef __attribute__((ext_vector_type(8)))  __bf16 v8bf;
typedef __attribute__((ext_vector_type(8)))  float  v8f;

__device__ __forceinline__ __bf16 f2bf(float x) {
    unsigned u = __builtin_bit_cast(unsigned, x);
    unsigned r = u + 0x7fffu + ((u >> 16) & 1u);   // round-to-nearest-even
    unsigned short h = (unsigned short)(r >> 16);
    return __builtin_bit_cast(__bf16, h);
}

// ---------------------------------------------------------------- converts
__global__ void mmp_f32_to_bf16(const float* __restrict__ in, __bf16* __restrict__ out, long n) {
    long i = (long)blockIdx.x * blockDim.x + threadIdx.x;
    long stride = (long)gridDim.x * blockDim.x;
    for (; i < n; i += stride) out[i] = f2bf(in[i]);
}

// ---------------------------------------------------------------- pattern inverse norms
__global__ void mmp_pattern_norms(const float* __restrict__ patterns, float* __restrict__ pinv) {
    int p = threadIdx.x;
    if (p >= NPAT) return;
    float s = 0.f;
    const float* row = patterns + (size_t)p * FDIM;
    for (int d = 0; d < FDIM; ++d) { float v = row[d]; s += v * v; }
    pinv[p] = 1.f / fmaxf(sqrtf(s), 1e-12f);
}

// ---------------------------------------------------------------- K/V projections of the 50 patterns
__global__ void mmp_pattern_kv(const float* __restrict__ patterns,
                               const float* __restrict__ in_proj_w,
                               const float* __restrict__ in_proj_b,
                               float* __restrict__ kpat, float* __restrict__ vpat) {
    __shared__ float sp[FDIM];
    int p = blockIdx.x;
    for (int d = threadIdx.x; d < FDIM; d += blockDim.x) sp[d] = patterns[(size_t)p * FDIM + d];
    __syncthreads();
    int n = blockIdx.y * blockDim.x + threadIdx.x;
    const float* wk = in_proj_w + (size_t)FDIM * FDIM     + (size_t)n * FDIM;
    const float* wv = in_proj_w + (size_t)2 * FDIM * FDIM + (size_t)n * FDIM;
    float sk = 0.f, sv = 0.f;
    for (int d = 0; d < FDIM; ++d) { float pv = sp[d]; sk += pv * wk[d]; sv += pv * wv[d]; }
    kpat[(size_t)p * FDIM + n] = sk + in_proj_b[FDIM + n];
    vpat[(size_t)p * FDIM + n] = sv + in_proj_b[2 * FDIM + n];
}

// ---------------------------------------------------------------- cosine sim + top-3 (one wave per row)
__global__ __launch_bounds__(256) void mmp_sim_topk(const float* __restrict__ x,
                                                    const float* __restrict__ patterns,
                                                    const float* __restrict__ pinv,
                                                    int* __restrict__ topidx) {
    int lane = threadIdx.x & 31;
    int row  = blockIdx.x * 8 + (threadIdx.x >> 5);
    const float* xr = x + (size_t)row * FDIM;

    float dot[NPAT];
#pragma unroll
    for (int p = 0; p < NPAT; ++p) dot[p] = 0.f;
    float xn = 0.f;
#pragma unroll
    for (int it = 0; it < FDIM / 128; ++it) {
        int d = it * 128 + lane * 4;
        float4 xv = *(const float4*)(xr + d);
        xn += xv.x * xv.x + xv.y * xv.y + xv.z * xv.z + xv.w * xv.w;
#pragma unroll
        for (int p = 0; p < NPAT; ++p) {
            float4 pv = *(const float4*)(patterns + (size_t)p * FDIM + d);
            dot[p] += xv.x * pv.x + xv.y * pv.y + xv.z * pv.z + xv.w * pv.w;
        }
    }
#pragma unroll
    for (int off = 16; off > 0; off >>= 1) xn += __shfl_xor(xn, off, 32);
#pragma unroll
    for (int p = 0; p < NPAT; ++p) {
        float v = dot[p];
#pragma unroll
        for (int off = 16; off > 0; off >>= 1) v += __shfl_xor(v, off, 32);
        dot[p] = v;
    }
    if (lane == 0) {
        float invx = 1.f / fmaxf(sqrtf(xn), 1e-12f);
        float sim[NPAT];
#pragma unroll
        for (int p = 0; p < NPAT; ++p) sim[p] = dot[p] * invx * pinv[p];
        int s0 = 0, s1 = 0, s2 = 0;
        float b = -1e30f;
#pragma unroll
        for (int p = 0; p < NPAT; ++p) if (sim[p] > b) { b = sim[p]; s0 = p; }
        b = -1e30f;
#pragma unroll
        for (int p = 0; p < NPAT; ++p) if (p != s0 && sim[p] > b) { b = sim[p]; s1 = p; }
        b = -1e30f;
#pragma unroll
        for (int p = 0; p < NPAT; ++p) if (p != s0 && p != s1 && sim[p] > b) { b = sim[p]; s2 = p; }
        topidx[row * 3 + 0] = s0;
        topidx[row * 3 + 1] = s1;
        topidx[row * 3 + 2] = s2;
    }
}

// ---------------------------------------------------------------- bf16 WMMA GEMM: out = A[M,K] * W[N,K]^T + bias (+res)(relu)
// block 256 = 8 waves (2 x 4), wave tile 64x64 (16 WMMA / 16 b128-loads per K-step),
// block tile 128x256. grid (N/256, M/128). K is compile-time so fragment strides
// become 24-bit immediate offsets on global_load_b128.
template <int K, bool RELU, bool RES, bool OUTBF>
__global__ __launch_bounds__(256) void mmp_gemm_bf16(const __bf16* __restrict__ A,
                                                     const __bf16* __restrict__ W,
                                                     const float* __restrict__ bias,
                                                     const float* __restrict__ res,
                                                     float* __restrict__ outF,
                                                     __bf16* __restrict__ outB,
                                                     int M, int N) {
    int lane = threadIdx.x & 31;
    int wave = threadIdx.x >> 5;
    int row0 = blockIdx.y * 128 + (wave >> 2) * 64;
    int col0 = blockIdx.x * 256 + (wave & 3) * 64;

    int l15 = lane & 15;
    int khA = (lane >> 4) << 3;   // A: lane half owns K {0-7,16-23} / {8-15,24-31}
    int khB = (lane >> 4) << 3;   // B: lane half owns K {0-7,16-23} / {8-15,24-31}
    const __bf16* pa = A + (size_t)(row0 + l15) * K + khA;
    const __bf16* pb = W + (size_t)(col0 + l15) * K + khB;

    v8f z = {0.f, 0.f, 0.f, 0.f, 0.f, 0.f, 0.f, 0.f};
    v8f acc[4][4];
#pragma unroll
    for (int i = 0; i < 4; ++i)
#pragma unroll
        for (int j = 0; j < 4; ++j) acc[i][j] = z;

    for (int k0 = 0; k0 < K; k0 += 32, pa += 32, pb += 32) {
        v16bf a[4], b[4];
#pragma unroll
        for (int i = 0; i < 4; ++i) {
            const __bf16* p = pa + i * (16 * K);          // constant stride -> imm offset
            v8bf lo = *(const v8bf*)(p);
            v8bf hi = *(const v8bf*)(p + 16);
#pragma unroll
            for (int t = 0; t < 8; ++t) { a[i][t] = lo[t]; a[i][t + 8] = hi[t]; }
        }
#pragma unroll
        for (int j = 0; j < 4; ++j) {
            const __bf16* p = pb + j * (16 * K);          // constant stride -> imm offset
            v8bf lo = *(const v8bf*)(p);
            v8bf hi = *(const v8bf*)(p + 16);
#pragma unroll
            for (int t = 0; t < 8; ++t) { b[j][t] = lo[t]; b[j][t + 8] = hi[t]; }
        }
#pragma unroll
        for (int i = 0; i < 4; ++i)
#pragma unroll
            for (int j = 0; j < 4; ++j)
                acc[i][j] = __builtin_amdgcn_wmma_f32_16x16x32_bf16(
                    false, a[i], false, b[j], (short)0, acc[i][j], false, false);
    }

    int lh = (lane >> 4) << 3;    // C/D: lanes 16-31 hold M+8
#pragma unroll
    for (int j = 0; j < 4; ++j) {
        int col = col0 + j * 16 + l15;
        float bv = bias[col];
#pragma unroll
        for (int i = 0; i < 4; ++i) {
#pragma unroll
            for (int r = 0; r < 8; ++r) {
                int row = row0 + i * 16 + lh + r;
                size_t o = (size_t)row * N + col;
                float v = acc[i][j][r] + bv;
                if (RES)  v += res[o];
                if (RELU) v = fmaxf(v, 0.f);
                if (OUTBF) outB[o] = f2bf(v);
                else       outF[o] = v;
            }
        }
    }
}

// ---------------------------------------------------------------- attention: one wave per (row, head)
__global__ __launch_bounds__(256) void mmp_attention(const float* __restrict__ Q,
                                                     const float* __restrict__ kpat,
                                                     const float* __restrict__ vpat,
                                                     const int* __restrict__ topidx,
                                                     __bf16* __restrict__ ctx,
                                                     float* __restrict__ attn_out) {
    __shared__ float sw[NHEAD][TOPK];
    int b    = blockIdx.x;
    int h    = threadIdx.x >> 5;
    int lane = threadIdx.x & 31;
    int i0 = topidx[b * 3 + 0], i1 = topidx[b * 3 + 1], i2 = topidx[b * 3 + 2];
    const float* q  = Q    + (size_t)b  * FDIM + h * 128;
    const float* k0 = kpat + (size_t)i0 * FDIM + h * 128;
    const float* k1 = kpat + (size_t)i1 * FDIM + h * 128;
    const float* k2 = kpat + (size_t)i2 * FDIM + h * 128;
    float s0 = 0.f, s1 = 0.f, s2 = 0.f;
#pragma unroll
    for (int j = 0; j < 4; ++j) {
        int d = lane + 32 * j;
        float qv = q[d];
        s0 += qv * k0[d]; s1 += qv * k1[d]; s2 += qv * k2[d];
    }
#pragma unroll
    for (int off = 16; off > 0; off >>= 1) {
        s0 += __shfl_xor(s0, off, 32);
        s1 += __shfl_xor(s1, off, 32);
        s2 += __shfl_xor(s2, off, 32);
    }
    const float scale = 0.08838834764831845f;  // 1/sqrt(128)
    s0 *= scale; s1 *= scale; s2 *= scale;
    float m  = fmaxf(s0, fmaxf(s1, s2));
    float e0 = __expf(s0 - m), e1 = __expf(s1 - m), e2 = __expf(s2 - m);
    float inv = 1.f / (e0 + e1 + e2);
    float a0 = e0 * inv, a1 = e1 * inv, a2 = e2 * inv;
    const float* v0 = vpat + (size_t)i0 * FDIM + h * 128;
    const float* v1 = vpat + (size_t)i1 * FDIM + h * 128;
    const float* v2 = vpat + (size_t)i2 * FDIM + h * 128;
#pragma unroll
    for (int j = 0; j < 4; ++j) {
        int d = lane + 32 * j;
        float c = a0 * v0[d] + a1 * v1[d] + a2 * v2[d];
        ctx[(size_t)b * FDIM + h * 128 + d] = f2bf(c);
    }
    if (lane == 0) { sw[h][0] = a0; sw[h][1] = a1; sw[h][2] = a2; }
    __syncthreads();
    if (threadIdx.x < TOPK) {
        float s = 0.f;
#pragma unroll
        for (int hh = 0; hh < NHEAD; ++hh) s += sw[hh][threadIdx.x];
        attn_out[(size_t)b * TOPK + threadIdx.x] = s * (1.f / NHEAD);
    }
}

// ---------------------------------------------------------------- launcher
extern "C" void kernel_launch(void* const* d_in, const int* in_sizes, int n_in,
                              void* d_out, int out_size, void* d_ws, size_t ws_size,
                              hipStream_t stream) {
    const float* x         = (const float*)d_in[0];
    const float* patterns  = (const float*)d_in[1];
    const float* in_proj_w = (const float*)d_in[2];
    const float* in_proj_b = (const float*)d_in[3];
    const float* out_w     = (const float*)d_in[4];
    const float* out_b     = (const float*)d_in[5];
    const float* w1        = (const float*)d_in[6];
    const float* b1        = (const float*)d_in[7];
    const float* w2        = (const float*)d_in[8];
    const float* b2        = (const float*)d_in[9];
    const float* w3        = (const float*)d_in[10];
    const float* b3        = (const float*)d_in[11];

    float* pred = (float*)d_out;
    float* attn = pred + (size_t)BATCH * FDIM;

    // workspace carve (256B aligned)
    char*  ws  = (char*)d_ws;
    size_t off = 0;
    auto carve = [&](size_t bytes) -> char* {
        char* p = ws + off;
        off = (off + bytes + 255) & ~(size_t)255;
        return p;
    };
    __bf16* x_bf    = (__bf16*)carve((size_t)BATCH * FDIM * 2);
    __bf16* wq_bf   = (__bf16*)carve((size_t)FDIM * FDIM * 2);
    __bf16* outw_bf = (__bf16*)carve((size_t)FDIM * FDIM * 2);
    __bf16* w1_bf   = (__bf16*)carve((size_t)HDIM * FDIM * 2);
    __bf16* w2_bf   = (__bf16*)carve((size_t)HDIM * HDIM * 2);
    __bf16* w3_bf   = (__bf16*)carve((size_t)FDIM * HDIM * 2);
    float*  kpat    = (float*)carve((size_t)NPAT * FDIM * 4);
    float*  vpat    = (float*)carve((size_t)NPAT * FDIM * 4);
    float*  pinv    = (float*)carve((size_t)NPAT * 4);
    int*    topidx  = (int*)carve((size_t)BATCH * TOPK * 4);
    float*  qbuf    = (float*)carve((size_t)BATCH * FDIM * 4);
    __bf16* ctx_bf  = (__bf16*)carve((size_t)BATCH * FDIM * 2);
    __bf16* xc_bf   = (__bf16*)carve((size_t)BATCH * FDIM * 2);
    __bf16* h1_bf   = (__bf16*)carve((size_t)BATCH * HDIM * 2);
    __bf16* h2_bf   = (__bf16*)carve((size_t)BATCH * HDIM * 2);

    // --- precision converts (f32 -> bf16) ---
    const int CB = 256, CG = 4096;
    mmp_f32_to_bf16<<<CG, CB, 0, stream>>>(x,         x_bf,    (long)BATCH * FDIM);
    mmp_f32_to_bf16<<<CG, CB, 0, stream>>>(in_proj_w, wq_bf,   (long)FDIM * FDIM);  // Wq = first D rows
    mmp_f32_to_bf16<<<CG, CB, 0, stream>>>(out_w,     outw_bf, (long)FDIM * FDIM);
    mmp_f32_to_bf16<<<CG, CB, 0, stream>>>(w1,        w1_bf,   (long)HDIM * FDIM);
    mmp_f32_to_bf16<<<CG, CB, 0, stream>>>(w2,        w2_bf,   (long)HDIM * HDIM);
    mmp_f32_to_bf16<<<CG, CB, 0, stream>>>(w3,        w3_bf,   (long)FDIM * HDIM);

    // --- pattern-side precompute (tiny) ---
    mmp_pattern_norms<<<1, 64, 0, stream>>>(patterns, pinv);
    mmp_pattern_kv<<<dim3(NPAT, FDIM / 128), 128, 0, stream>>>(patterns, in_proj_w, in_proj_b, kpat, vpat);

    // --- cosine similarity + top-3 ---
    mmp_sim_topk<<<BATCH / 8, 256, 0, stream>>>(x, patterns, pinv, topidx);

    // --- Q = x @ Wq^T + bq (f32 out) ---
    mmp_gemm_bf16<FDIM, false, false, false><<<dim3(FDIM / 256, BATCH / 128), 256, 0, stream>>>(
        x_bf, wq_bf, in_proj_b, nullptr, qbuf, nullptr, BATCH, FDIM);

    // --- attention over gathered K_pat/V_pat -> ctx (bf16) + attn weights ---
    mmp_attention<<<BATCH, 256, 0, stream>>>(qbuf, kpat, vpat, topidx, ctx_bf, attn);

    // --- xc = x + ctx @ out_w^T + out_b (bf16 out) ---
    mmp_gemm_bf16<FDIM, false, true, true><<<dim3(FDIM / 256, BATCH / 128), 256, 0, stream>>>(
        ctx_bf, outw_bf, out_b, x, nullptr, xc_bf, BATCH, FDIM);

    // --- h1 = relu(xc @ W1^T + b1) ---
    mmp_gemm_bf16<FDIM, true, false, true><<<dim3(HDIM / 256, BATCH / 128), 256, 0, stream>>>(
        xc_bf, w1_bf, b1, nullptr, nullptr, h1_bf, BATCH, HDIM);

    // --- h2 = relu(h1 @ W2^T + b2) ---
    mmp_gemm_bf16<HDIM, true, false, true><<<dim3(HDIM / 256, BATCH / 128), 256, 0, stream>>>(
        h1_bf, w2_bf, b2, nullptr, nullptr, h2_bf, BATCH, HDIM);

    // --- pred = x + h2 @ W3^T + b3 (f32 out to d_out) ---
    mmp_gemm_bf16<HDIM, false, true, false><<<dim3(FDIM / 256, BATCH / 128), 256, 0, stream>>>(
        h2_bf, w3_bf, b3, x, pred, nullptr, BATCH, FDIM);
}
